// NetVLAD_32349693674036
// MI455X (gfx1250) — matrix operand order repaired and checked by reading, not compiled, not run on hardware
//
#include <hip/hip_runtime.h>

typedef __attribute__((ext_vector_type(16))) __bf16 v16bf;
typedef __attribute__((ext_vector_type(8)))  __bf16 v8bf;
typedef __attribute__((ext_vector_type(8)))  float  v8f;

#define C_DIM 128
#define K_DIM 64
#define S_DIM 4096
#define N_IMG 128
#define SC    256
#define NCHUNK (S_DIM / SC)

// LDS element strides
#define XF_STR  272   // xf  [C][XF_STR]   bf16 (544 B rows, 32B aligned)
#define CT_STR  144   // xfT [SC][CT_STR]  bf16 (288 B rows, 32B aligned)
#define LGT_STR 68    // lgT [SC][LGT_STR] f32  (272 B rows, 16B aligned)

// byte offsets into dynamic LDS
#define OFF_XF   0
#define OFF_XFT  (OFF_XF  + C_DIM * XF_STR * 2)      // 69632
#define OFF_WSW  (OFF_XFT + SC * CT_STR * 2)         // 143360
#define OFF_ASW  (OFF_WSW + K_DIM * C_DIM * 2)       // 159744
#define OFF_LGT  (OFF_ASW + K_DIM * SC * 2)          // 192512
#define OFF_BIAS (OFF_LGT + SC * LGT_STR * 4)        // 262144
#define OFF_INV  (OFF_BIAS + 256)                    // 262400
#define SMEM1_BYTES (OFF_INV + SC * 4)               // 263424

__global__ void zero_ws_kernel(float* __restrict__ p, int nElem) {
    int i = blockIdx.x * blockDim.x + threadIdx.x;
    if (i < nElem) p[i] = 0.0f;
}

// swizzled A-fragment index: row m, contraction index kidx within a 32-wide step
// half-wave holds koff {0..15} (half=0) / {16..31} (half=1)
__device__ __forceinline__ int a_swz(int step, int m, int kidx32) {
    int half = (kidx32 >> 4) & 1;
    int e    = kidx32 & 15;
    return ((step * K_DIM + m) * 2 + half) * 16 + e;
}

__global__ void __launch_bounds__(256)
netvlad_phase1(const float* __restrict__ x, const float* __restrict__ conv_w,
               const float* __restrict__ conv_b, float* __restrict__ vlad_ws,
               float* __restrict__ asum_ws)
{
    extern __shared__ __align__(32) char smem[];
    __bf16* xf   = (__bf16*)(smem + OFF_XF);    // [C][XF_STR]  normalized, GEMM2 B
    __bf16* xfT  = (__bf16*)(smem + OFF_XFT);   // [SC][CT_STR] normalized, GEMM1 B
    __bf16* wsw  = (__bf16*)(smem + OFF_WSW);   // W, swizzled A-fragment order
    __bf16* asw  = (__bf16*)(smem + OFF_ASW);   // softmax a, swizzled A-fragment order
    float*  lgT  = (float*) (smem + OFF_LGT);   // logits transposed [SC][LGT_STR]
    float*  bias = (float*) (smem + OFF_BIAS);  // [K]
    float*  invs = (float*) (smem + OFF_INV);   // per-column 1/sum [SC]

    const int t    = threadIdx.x;
    const int lane = t & 31;
    const int wv   = t >> 5;          // wave id 0..7
    const int hi   = (lane >> 4) & 1; // half-wave
    const int ln   = lane & 15;
    const int n    = blockIdx.y;
    const int s0   = blockIdx.x * SC;

    // ---- stage W (f32 -> bf16) directly in swizzled A-fragment layout ----
    for (int i = t; i < K_DIM * C_DIM; i += 256) {
        int k = i >> 7, c = i & 127;
        wsw[a_swz(c >> 5, k, c & 31)] = (__bf16)conv_w[i];
    }
    if (t < K_DIM) bias[t] = conv_b[t];

    // ---- pass 1: read x once, accumulate column sum-of-squares,
    //              park raw bf16 in xfT (vectorized 16B LDS stores) ----
    const float* xg = x + (size_t)n * C_DIM * S_DIM + s0;
    float ssq = 0.0f;
    for (int c8 = 0; c8 < C_DIM; c8 += 8) {
        v8bf pk;
        #pragma unroll
        for (int j = 0; j < 8; ++j) {
            float v = xg[(size_t)(c8 + j) * S_DIM + t];
            ssq += v * v;
            pk[j] = (__bf16)v;
        }
        *(v8bf*)(xfT + t * CT_STR + c8) = pk;
    }
    float nscale = 1.0f / fmaxf(sqrtf(ssq), 1e-12f);

    // ---- pass 2: rescale from LDS; refresh xfT and scatter into xf[c][s] ----
    for (int c8 = 0; c8 < C_DIM; c8 += 8) {
        v8bf pk = *(v8bf*)(xfT + t * CT_STR + c8);
        #pragma unroll
        for (int j = 0; j < 8; ++j) {
            __bf16 v = (__bf16)((float)pk[j] * nscale);
            pk[j] = v;
            xf[(c8 + j) * XF_STR + t] = v;
        }
        *(v8bf*)(xfT + t * CT_STR + c8) = pk;
    }
    __syncthreads();

    // ---- GEMM1: logits[64, SC] = W[64,128] @ xf[128, SC]
    // Each wave owns one M-tile (mt = wv>>1) and 8 N-tiles.
    {
        const int mt = wv >> 1;
        v16bf aG[4];
        #pragma unroll
        for (int kk = 0; kk < 4; ++kk)
            aG[kk] = *(const v16bf*)(wsw + ((kk * K_DIM + mt * 16 + ln) * 2 + hi) * 16);
        v8f bv = *(const v8f*)(bias + mt * 16 + hi * 8);

        for (int i = 0; i < 8; ++i) {
            int nt = (wv & 1) * 8 + i;
            v8f acc = {};
            #pragma unroll
            for (int kk = 0; kk < 4; ++kk) {
                v16bf b = *(const v16bf*)(xfT + (nt * 16 + ln) * CT_STR + kk * 32 + hi * 16);
                acc = __builtin_amdgcn_wmma_f32_16x16x32_bf16(
                    false, aG[kk], false, b, (short)0, acc, false, false);
            }
            // D rows for this lane are contiguous: mt*16 + hi*8 + (0..7)
            *(v8f*)(lgT + (nt * 16 + ln) * LGT_STR + mt * 16 + hi * 8) = acc + bv;
        }
    }
    __syncthreads();

    // ---- softmax over K=64 for column t (contiguous row of lgT) ----
    {
        float* row = lgT + t * LGT_STR;
        float m = -3.4e38f;
        for (int k = 0; k < K_DIM; ++k) m = fmaxf(m, row[k]);
        float sum = 0.0f;
        for (int k = 0; k < K_DIM; ++k) {
            float e = __expf(row[k] - m);
            row[k] = e;
            sum += e;
        }
        float inv = 1.0f / sum;
        invs[t] = inv;
        const int swbase = a_swz(t >> 5, 0, t & 31);   // + k*32 per row
        for (int k = 0; k < K_DIM; ++k)
            asw[swbase + k * 32] = (__bf16)(row[k] * inv);
    }
    __syncthreads();

    // ---- per-chunk asum[k] = sum_s a[k,s] ----
    if (t < K_DIM) {
        float s = 0.0f;
        for (int j = 0; j < SC; ++j) s += lgT[j * LGT_STR + t] * invs[j];
        atomicAdd(&asum_ws[n * K_DIM + t], s);
    }

    // ---- GEMM2: vlad[64,128] += a[64,SC] @ xf^T[SC,128]
    // Each wave owns one M-tile (mt = wv>>1) and 4 N-tiles.
    {
        const int mt = wv >> 1;
        v16bf aG[8];
        #pragma unroll
        for (int ks = 0; ks < 8; ++ks)
            aG[ks] = *(const v16bf*)(asw + ((ks * K_DIM + mt * 16 + ln) * 2 + hi) * 16);

        for (int i = 0; i < 4; ++i) {
            int nt = (wv & 1) * 4 + i;
            v8f acc = {};
            #pragma unroll
            for (int ks = 0; ks < 8; ++ks) {
                v16bf b = *(const v16bf*)(xf + (nt * 16 + ln) * XF_STR + ks * 32 + hi * 16);
                acc = __builtin_amdgcn_wmma_f32_16x16x32_bf16(
                    false, aG[ks], false, b, (short)0, acc, false, false);
            }
            float* dst = vlad_ws + ((size_t)n * K_DIM + mt * 16 + hi * 8) * C_DIM
                                 + nt * 16 + ln;
            #pragma unroll
            for (int r = 0; r < 8; ++r)
                atomicAdd(dst + (size_t)r * C_DIM, acc[r]);
        }
    }
}

__global__ void __launch_bounds__(256)
netvlad_phase2(const float* __restrict__ vlad_ws, const float* __restrict__ asum_ws,
               const float* __restrict__ centroids, float* __restrict__ out)
{
    __shared__ float sv[K_DIM][132];
    __shared__ float rp[K_DIM][4];
    __shared__ float rsc[K_DIM];
    __shared__ float red[256];

    const int n = blockIdx.x;
    const int t = threadIdx.x;
    const int k = t >> 2, part = t & 3, c0 = part * 32;

    float ak = asum_ws[n * K_DIM + k];
    const float* vsrc = vlad_ws + ((size_t)n * K_DIM + k) * C_DIM;
    const float* cent = centroids + k * C_DIM;

    float ps = 0.0f;
    for (int i = 0; i < 32; ++i) {
        int c = c0 + i;
        float v = vsrc[c] - ak * cent[c];
        sv[k][c] = v;
        ps += v * v;
    }
    rp[k][part] = ps;
    __syncthreads();

    if (t < K_DIM) {
        float s = rp[t][0] + rp[t][1] + rp[t][2] + rp[t][3];
        rsc[t] = 1.0f / fmaxf(sqrtf(s), 1e-12f);
    }
    __syncthreads();

    float sc = rsc[k];
    float ps2 = 0.0f;
    for (int i = 0; i < 32; ++i) {
        int c = c0 + i;
        float v = sv[k][c] * sc;
        sv[k][c] = v;
        ps2 += v * v;
    }
    red[t] = ps2;
    __syncthreads();
    for (int s = 128; s > 0; s >>= 1) {
        if (t < s) red[t] += red[t + s];
        __syncthreads();
    }
    float tsc = 1.0f / fmaxf(sqrtf(red[0]), 1e-12f);

    float* o = out + (size_t)n * (K_DIM * C_DIM) + k * C_DIM;
    for (int i = 0; i < 32; ++i) o[c0 + i] = sv[k][c0 + i] * tsc;
}

extern "C" void kernel_launch(void* const* d_in, const int* in_sizes, int n_in,
                              void* d_out, int out_size, void* d_ws, size_t ws_size,
                              hipStream_t stream) {
    (void)in_sizes; (void)n_in; (void)out_size; (void)ws_size;
    const float* x         = (const float*)d_in[0];
    const float* conv_w    = (const float*)d_in[1];
    const float* conv_b    = (const float*)d_in[2];
    const float* centroids = (const float*)d_in[3];
    float* out = (float*)d_out;

    float* vlad_ws = (float*)d_ws;                                  // [N,K,C] f32
    float* asum_ws = vlad_ws + (size_t)N_IMG * K_DIM * C_DIM;       // [N,K]   f32

    hipFuncSetAttribute((const void*)netvlad_phase1,
                        hipFuncAttributeMaxDynamicSharedMemorySize, SMEM1_BYTES);

    int ztot = N_IMG * K_DIM * C_DIM + N_IMG * K_DIM;
    zero_ws_kernel<<<(ztot + 255) / 256, 256, 0, stream>>>(vlad_ws, ztot);

    dim3 g1(NCHUNK, N_IMG);
    netvlad_phase1<<<g1, 256, SMEM1_BYTES, stream>>>(x, conv_w, conv_b,
                                                     vlad_ws, asum_ws);

    netvlad_phase2<<<N_IMG, 256, 0, stream>>>(vlad_ws, asum_ws, centroids, out);
}
